// GraphConvWat_35716948034099
// MI455X (gfx1250) — matrix operand order repaired and checked by reading, not compiled, not run on hardware
//
#include <hip/hip_runtime.h>
#include <cstdint>
#include <cstddef>

typedef __attribute__((ext_vector_type(16))) _Float16 v16h;
typedef __attribute__((ext_vector_type(8)))  float    v8f;

#define WG 256

__device__ __forceinline__ void fatomic_add(float* p, float v) {
  (void)__hip_atomic_fetch_add(p, v, __ATOMIC_RELAXED, __HIP_MEMORY_SCOPE_AGENT);
}

// ---------------- utility kernels ----------------

__global__ void k_zero4(float4* __restrict__ p, int count4) {
  int i = blockIdx.x * blockDim.x + threadIdx.x;
  if (i < count4) p[i] = make_float4(0.f, 0.f, 0.f, 0.f);
}

__global__ void k_deg(const int* __restrict__ src, float* __restrict__ deg, int E) {
  int i = blockIdx.x * blockDim.x + threadIdx.x;
  if (i < E) fatomic_add(&deg[src[i]], 1.0f);
}

__global__ void k_wnorm(const int* __restrict__ src, const int* __restrict__ dst,
                        const float* __restrict__ deg, float* __restrict__ w, int E) {
  int i = blockIdx.x * blockDim.x + threadIdx.x;
  if (i >= E) return;
  float ds = deg[src[i]], dd = deg[dst[i]];
  float a = (ds > 0.f) ? rsqrtf(ds) : 0.f;
  float b = (dd > 0.f) ? rsqrtf(dd) : 0.f;
  w[i] = -a * b;   // -D^-1/2 A D^-1/2  (lambda_max = 2 => L_hat)
}

// pack [n, xdim] rows (source row stride xs) into stride-tstr buffer, zero pad cols
__global__ void k_pack(const float* __restrict__ X, float* __restrict__ T,
                       int n, int xs, int xdim, int tstr) {
  int idx = blockIdx.x * blockDim.x + threadIdx.x;
  if (idx >= n * tstr) return;
  int i = idx / tstr, j = idx - i * tstr;
  T[idx] = (j < xdim) ? X[(size_t)i * xs + j] : 0.f;
}

// flat B = -A over count4 float4s (T slots are contiguous now)
__global__ void k_negcopy(const float4* __restrict__ A, float4* __restrict__ B, int count4) {
  int idx = blockIdx.x * blockDim.x + threadIdx.x;
  if (idx >= count4) return;
  float4 v = A[idx];
  B[idx] = make_float4(-v.x, -v.y, -v.z, -v.w);
}

// Tout[dst] += factor * w[e] * Tin[src], 4 features per thread (blockIdx.y = chunk)
__global__ void k_scatter(const int* __restrict__ src, const int* __restrict__ dst,
                          const float* __restrict__ w, const float* __restrict__ Tin,
                          float* __restrict__ Tout, int E, float factor, int str) {
  int i = blockIdx.x * blockDim.x + threadIdx.x;
  if (i >= E) return;
  // stream-ahead prefetch of the edge arrays (speculative, OOB-safe)
  __builtin_prefetch(&w[i + 8 * WG], 0, 1);
  __builtin_prefetch(&src[i + 8 * WG], 0, 1);
  __builtin_prefetch(&dst[i + 8 * WG], 0, 1);
  int c = blockIdx.y << 2;
  float wf = w[i] * factor;
  int s = src[i], d = dst[i];
  const float4 v = *(const float4*)(Tin + (size_t)s * str + c);
  float* o = Tout + (size_t)d * str + c;
  fatomic_add(o + 0, wf * v.x);
  fatomic_add(o + 1, wf * v.y);
  fatomic_add(o + 2, wf * v.z);
  fatomic_add(o + 3, wf * v.w);
}

// Convert W [K, inDim, outDim] f32 -> fragment-ordered, zero-padded f16:
// WF[((k*inChunks+kc)*outTiles+ot)*512 + lane*16 + i] = B-fragment element
// per CDNA5 16-bit B layout: lane<16 -> K=kc*32+(i<8 ? i : 8+i),    N=ot*16+lane
//                            lane>=16 -> K=kc*32+8+(i<8 ? i : 8+i), N=ot*16+lane-16
__global__ void k_convw(const float* __restrict__ W, _Float16* __restrict__ WF,
                        int inDim, int outDim, int inChunks, int outTiles, int total) {
  int idx = blockIdx.x * blockDim.x + threadIdx.x;
  if (idx >= total) return;
  int i    = idx & 15;
  int lane = (idx >> 4) & 31;
  int rest = idx >> 9;
  int ot = rest % outTiles; rest /= outTiles;
  int kc = rest % inChunks;
  int k  = rest / inChunks;
  int kk = kc * 32 + ((lane < 16) ? 0 : 8) + ((i < 8) ? i : 8 + i);
  int nn = ot * 16 + (lane & 15);
  float v = (kk < inDim && nn < outDim) ? W[((size_t)k * inDim + kk) * outDim + nn] : 0.f;
  WF[idx] = (_Float16)v;
}

// ---------------- grouped WMMA GEMM-accumulate ----------------
// One wave per 16-row slab. Loads all OT 16x16 C fragments ONCE, then loops over
// g consecutive T_k ring slots, issuing IC*OT WMMAs per slot, and stores C once:
//   ACC[slab,:] += sum_j  T_{k0+j}[slab,:] @ W16[k0+j]
template<int IC, int OT, int NR>
__global__ __launch_bounds__(32) void k_gemm_g(const float* __restrict__ ring,
                                               long slotFloats, int startSlot, int g,
                                               const _Float16* __restrict__ WFg,
                                               float* __restrict__ ACC) {
  constexpr int ISTR = IC * 32;
  constexpr int OSTR = OT * 16;
  int lane = threadIdx.x;
  int tm = blockIdx.x;
  int hi = lane >> 4;                    // 0: lanes 0-15, 1: lanes 16-31
  int colL = lane & 15;
  long rowC = (long)tm * 16 + (hi ? 8 : 0);

  v8f c[OT];
#pragma unroll
  for (int ot = 0; ot < OT; ++ot)
#pragma unroll
    for (int r = 0; r < 8; ++r)
      c[ot][r] = ACC[(rowC + r) * OSTR + ot * 16 + colL];

  long m = (long)tm * 16 + colL;
  for (int j = 0; j < g; ++j) {
    int slot = startSlot + j; if (slot >= NR) slot -= NR;
    const float* arow = ring + (long)slot * slotFloats + m * ISTR + (hi ? 8 : 0);
    const _Float16* WFk = WFg + (long)j * (IC * OT * 512);
#pragma unroll
    for (int kc = 0; kc < IC; ++kc) {
      const float* ar = arow + kc * 32;
      float4 p0 = *(const float4*)(ar);        // K offsets +0..3
      float4 p1 = *(const float4*)(ar + 4);    // +4..7
      float4 p2 = *(const float4*)(ar + 16);   // +16..19
      float4 p3 = *(const float4*)(ar + 20);   // +20..23
      v16h a;
      a[0]  = (_Float16)p0.x; a[1]  = (_Float16)p0.y; a[2]  = (_Float16)p0.z; a[3]  = (_Float16)p0.w;
      a[4]  = (_Float16)p1.x; a[5]  = (_Float16)p1.y; a[6]  = (_Float16)p1.z; a[7]  = (_Float16)p1.w;
      a[8]  = (_Float16)p2.x; a[9]  = (_Float16)p2.y; a[10] = (_Float16)p2.z; a[11] = (_Float16)p2.w;
      a[12] = (_Float16)p3.x; a[13] = (_Float16)p3.y; a[14] = (_Float16)p3.z; a[15] = (_Float16)p3.w;

#pragma unroll
      for (int ot = 0; ot < OT; ++ot) {
        v16h b = *(const v16h*)(WFk + ((long)kc * OT + ot) * 512 + lane * 16);
        c[ot] = __builtin_amdgcn_wmma_f32_16x16x32_f16(false, a, false, b,
                                                       (short)0, c[ot], false, false);
      }
    }
  }
#pragma unroll
  for (int ot = 0; ot < OT; ++ot)
#pragma unroll
    for (int r = 0; r < 8; ++r)
      ACC[(rowC + r) * OSTR + ot * 16 + colL] = c[ot][r];
}

__global__ void k_bias_silu(const float* __restrict__ ACC, const float* __restrict__ bias,
                            float* __restrict__ H, int n, int ostr, int outDim) {
  int idx = blockIdx.x * blockDim.x + threadIdx.x;
  if (idx >= n * ostr) return;
  int j = idx % ostr;
  float v = ACC[idx];
  if (j < outDim) v += bias[j];
  H[idx] = v / (1.f + __expf(-v));   // silu; pad cols: silu(0)=0
}

__global__ void k_final(const float* __restrict__ H, const float* __restrict__ W4,
                        float* __restrict__ out, int n, int hstr, int inDim) {
  int i = blockIdx.x * blockDim.x + threadIdx.x;
  if (i >= n) return;
  const float* h = H + (size_t)i * hstr;
  float s = 0.f;
  for (int j = 0; j < inDim; ++j) s += h[j] * W4[j];
  out[i] = s;
}

// ---------------- host orchestration ----------------

static inline int cdiv(int a, int b) { return (a + b - 1) / b; }

// One ChebConv layer: ring-buffered recurrence + grouped WMMA accumulation.
template<int IC, int OT, int G>   // ring size NR == G (stream order makes this safe)
static void run_cheb(hipStream_t s, const int* esrc, const int* edst, const float* wn,
                     int E, int n, const float* Xsrc, int xStride, int xdim,
                     const float* W, const float* bias, int K, int inDim, int outDim,
                     float* ring, float* ACC, float* H, _Float16* WF) {
  constexpr int ISTR = IC * 32;
  constexpr int OSTR = OT * 16;
  constexpr int NR = G;
  const long slotFloats = (long)n * ISTR;
  const int slot4 = (int)(slotFloats / 4);
  const int chunksProp = cdiv(inDim, 4);
  const int gg = n / 16;                       // n is a multiple of 16

  k_pack<<<cdiv(n * ISTR, WG), WG, 0, s>>>(Xsrc, ring, n, xStride, xdim, ISTR); // slot0 = T0
  const int wtot = K * IC * OT * 512;
  k_convw<<<cdiv(wtot, WG), WG, 0, s>>>(W, WF, inDim, outDim, IC, OT, wtot);
  k_zero4<<<cdiv(n * OSTR / 4, WG), WG, 0, s>>>((float4*)ACC, n * OSTR / 4);

  if (K >= 2) {
    dim3 gs(cdiv(E, WG), chunksProp);
    k_zero4<<<cdiv(slot4, WG), WG, 0, s>>>((float4*)(ring + slotFloats), slot4); // slot1 = 0
    k_scatter<<<gs, WG, 0, s>>>(esrc, edst, wn, ring, ring + slotFloats, E, 1.0f, ISTR);
    k_gemm_g<IC, OT, NR><<<gg, 32, 0, s>>>(ring, slotFloats, 0, 2, WF, ACC);   // k = 0,1
    int k = 2;
    while (k < K) {
      int g = (K - k < G) ? (K - k) : G;
      int k0 = k;
      for (int j = 0; j < g; ++j, ++k) {
        int sk = k % NR, s1 = (k - 1) % NR, s2 = (k - 2) % NR;
        k_negcopy<<<cdiv(slot4, WG), WG, 0, s>>>(
            (const float4*)(ring + (long)s2 * slotFloats),
            (float4*)(ring + (long)sk * slotFloats), slot4);            // T_k = -T_{k-2}
        k_scatter<<<gs, WG, 0, s>>>(esrc, edst, wn,
            ring + (long)s1 * slotFloats,
            ring + (long)sk * slotFloats, E, 2.0f, ISTR);               // += 2 L_hat T_{k-1}
      }
      k_gemm_g<IC, OT, NR><<<gg, 32, 0, s>>>(ring, slotFloats, k0 % NR, g,
                                             WF + (long)k0 * IC * OT * 512, ACC);
    }
  } else {
    k_gemm_g<IC, OT, NR><<<gg, 32, 0, s>>>(ring, slotFloats, 0, 1, WF, ACC);
  }
  k_bias_silu<<<cdiv(n * OSTR, WG), WG, 0, s>>>(ACC, bias, H, n, OSTR, outDim);
}

extern "C" void kernel_launch(void* const* d_in, const int* in_sizes, int n_in,
                              void* d_out, int out_size, void* d_ws, size_t ws_size,
                              hipStream_t stream) {
  const float* x  = (const float*)d_in[0];
  // d_in[1] = edge_attr (unused by reference)
  const float* W1 = (const float*)d_in[2];
  const float* b1 = (const float*)d_in[3];
  const float* W2 = (const float*)d_in[4];
  const float* b2 = (const float*)d_in[5];
  const float* W3 = (const float*)d_in[6];
  const float* b3 = (const float*)d_in[7];
  const float* W4 = (const float*)d_in[8];
  const int*   ei = (const int*)d_in[9];
  // d_in[10] = batch (unused)

  const int n = in_sizes[0] / 3;       // 50000
  const int E = in_sizes[9] / 2;       // 800000
  const int* esrc = ei;
  const int* edst = ei + E;

  float* ws = (float*)d_ws;
  size_t off = 0;
  float* deg = ws + off; off += (size_t)n;       off = (off + 7) & ~(size_t)7;
  float* wn  = ws + off; off += (size_t)E;       off = (off + 7) & ~(size_t)7;
  float* ACC = ws + off; off += (size_t)n * 128;            // max OSTR = 128
  float* H   = ws + off; off += (size_t)n * 128;            // max OSTR = 128
  _Float16* WF = (_Float16*)(ws + off); off += (1u << 19);  // 1M halves (512K floats), 32B aligned
  float* ring = ws + off;
  // ring pool: max over layers of NR*n*ISTR floats:
  //   L1: 8*n*32 = 12.8M   L2: 4*n*128 = 25.6M   L3: 6*n*64 = 19.2M  -> 25.6M floats

  // normalized edge weights
  k_zero4<<<cdiv((n + 3) / 4, WG), WG, 0, stream>>>((float4*)deg, (n + 3) / 4);
  k_deg  <<<cdiv(E, WG), WG, 0, stream>>>(esrc, deg, E);
  k_wnorm<<<cdiv(E, WG), WG, 0, stream>>>(esrc, edst, deg, wn, E);

  // L1: K=240 in=3   out=120  (IC=1, OT=8, G=8)
  run_cheb<1, 8, 8>(stream, esrc, edst, wn, E, n, x, 3, 3,
                    W1, b1, 240, 3, 120, ring, ACC, H, WF);
  // L2: K=120 in=120 out=60   (IC=4, OT=4, G=4)
  run_cheb<4, 4, 4>(stream, esrc, edst, wn, E, n, H, 128, 120,
                    W2, b2, 120, 120, 60, ring, ACC, H, WF);
  // L3: K=20  in=60  out=30   (IC=2, OT=2, G=6)
  run_cheb<2, 2, 6>(stream, esrc, edst, wn, E, n, H, 64, 60,
                    W3, b3, 20, 60, 30, ring, ACC, H, WF);
  // L4: K=1   in=30  out=1, no bias/silu
  k_final<<<cdiv(n, WG), WG, 0, stream>>>(H, W4, (float*)d_out, n, 32, 30);
}
